// TransformerDecoderLayerWithMoE_60928406061078
// MI455X (gfx1250) — compile-verified, matrix operand order checked by reading, not run on hardware
//
#include <hip/hip_runtime.h>

// ---------------- types ----------------
typedef __bf16 bf16_t;
typedef __attribute__((ext_vector_type(16))) __bf16 v16bf;
typedef __attribute__((ext_vector_type(8)))  float  v8f;

#define B_   4
#define S_   2048
#define D_   1024
#define H_   16
#define E_   8
#define HID_ 4096
#define T_   (B_ * S_)
#define TK_  (2 * T_)

// ---------------- helpers ----------------
__device__ __forceinline__ bf16_t f2bf(float f) {
  unsigned int u = __builtin_bit_cast(unsigned int, f);
  unsigned int r = (u + 0x7FFFu + ((u >> 16) & 1u)) >> 16;
  unsigned short s = (unsigned short)r;
  return __builtin_bit_cast(bf16_t, s);
}

__device__ __forceinline__ v8f v8f_zero() {
  v8f z;
#pragma unroll
  for (int i = 0; i < 8; ++i) z[i] = 0.f;
  return z;
}

__device__ __forceinline__ v8f wmma_bf16(v16bf a, v16bf b, v8f c) {
  return __builtin_amdgcn_wmma_f32_16x16x32_bf16(false, a, false, b, (short)0, c,
                                                 false, false);
}

// CDNA5 async memory->LDS DMA (16 bytes per enabled lane), ASYNCcnt-tracked.
__device__ __forceinline__ void async_load_b128(const void* gaddr, void* lds) {
  unsigned ldsoff = (unsigned)(unsigned long long)lds;  // LDS byte address
  unsigned long long ga = (unsigned long long)gaddr;
  asm volatile("global_load_async_to_lds_b128 %0, %1, off"
               :
               : "v"(ldsoff), "v"(ga)
               : "memory");
}
__device__ __forceinline__ void wait_async0() {
  asm volatile("s_wait_asynccnt 0" ::: "memory");
}

// A-operand fragment (16x32 bf16 tile). t points at (row0, k0); ld = row stride (bf16).
// Per ISA: lane L (L<16) row M=L holds K={0..7,16..23}; lane L+16 holds K={8..15,24..31}.
__device__ __forceinline__ v16bf load_frag_a(const bf16_t* t, int ld) {
  int lane = threadIdx.x & 31;
  int m = lane & 15, half = (lane >> 4) & 1;
  const bf16_t* r = t + m * ld;
  v16bf f;
#pragma unroll
  for (int e = 0; e < 8; ++e) f[e] = r[half * 8 + e];
#pragma unroll
  for (int e = 0; e < 8; ++e) f[8 + e] = r[16 + half * 8 + e];
  return f;
}

// B-operand fragment (32x16 bf16 tile stored [k][n]); ld = k-row stride (bf16).
__device__ __forceinline__ v16bf load_frag_b(const bf16_t* t, int ld) {
  int lane = threadIdx.x & 31;
  int n = lane & 15, half = (lane >> 4) & 1;
  v16bf f;
#pragma unroll
  for (int e = 0; e < 8; ++e) f[e] = t[(half * 8 + e) * ld + n];
#pragma unroll
  for (int e = 0; e < 8; ++e) f[8 + e] = t[(16 + half * 8 + e) * ld + n];
  return f;
}

// ---------------- fp32 -> bf16 convert ----------------
__global__ void cvt_kernel(const float* __restrict__ src, bf16_t* __restrict__ dst,
                           long n) {
  long i = (long)blockIdx.x * blockDim.x + threadIdx.x;
  if (i < n) dst[i] = f2bf(src[i]);
}

// ---------------- tiled bf16 WMMA GEMM (double-buffered async LDS) ----------------
// C[M,N] = act(A[M,K] * W[K,N] + bias). 256 threads = 8 waves; tile 128x64x32.
// offs!=null: dynamic expert mode; rows = offs[e+1]-offs[e]; C row = offs[e]+r;
// rowidx!=null: A row = rowidx[offs[e]+r] (token gather), else A row = offs[e]+r.
#define BM 128
#define BN 64
#define BK 32
#define LDA_S 40
#define LDW_S 72

template <bool RELU, bool BF16OUT>
__global__ __launch_bounds__(256) void gemm_kernel(
    const bf16_t* __restrict__ A, const bf16_t* __restrict__ W,
    const float* __restrict__ bias, void* __restrict__ Cv,
    const int* __restrict__ offs, int eidx, const int* __restrict__ rowidx,
    int M, int N, int K, int lda, int ldc) {
  __shared__ bf16_t As[2][BM * LDA_S];
  __shared__ bf16_t Ws[2][BK * LDW_S];

  int off_e = 0, cnt = M;
  if (offs) { off_e = offs[eidx]; cnt = offs[eidx + 1] - off_e; }
  int row0 = blockIdx.y * BM;
  if (row0 >= cnt) return;
  int n0 = blockIdx.x * BN;

  int tid = threadIdx.x;
  int lane = tid & 31, wid = tid >> 5;
  int wm = (wid & 3) * 32, wn = (wid >> 2) * 32;

  // A staging: 2 threads per tile row, 16 bf16 (two async b128) each
  int ar_local = tid >> 1;
  int aseg = (tid & 1) * 16;
  int arow = row0 + ar_local;
  const bf16_t* aptr = nullptr;
  if (arow < cnt) {
    int g = rowidx ? rowidx[off_e + arow] : (off_e + arow);
    aptr = A + (long)g * lda;
  }
  // W staging: thread loads 8 bf16 (one async b128)
  int wkk = tid >> 3, wnc = (tid & 7) * 8;

  auto stage = [&](int k0s, int bufs) {
    if (aptr) {
      const bf16_t* g0 = aptr + k0s + aseg;
      bf16_t* l0 = &As[bufs][ar_local * LDA_S + aseg];
      async_load_b128(g0, l0);
      async_load_b128(g0 + 8, l0 + 8);
    }
    const bf16_t* g1 = W + (long)(k0s + wkk) * N + n0 + wnc;
    async_load_b128(g1, &Ws[bufs][wkk * LDW_S + wnc]);
  };

  v8f acc[2][2];
#pragma unroll
  for (int i = 0; i < 2; ++i)
#pragma unroll
    for (int j = 0; j < 2; ++j) acc[i][j] = v8f_zero();

  // software pipeline: DMA for tile i+1 overlaps WMMA compute on tile i
  stage(0, 0);
  wait_async0();
  __syncthreads();

  for (int k0 = 0; k0 < K; k0 += BK) {
    int buf = (k0 / BK) & 1;
    if (k0 + BK < K) stage(k0 + BK, buf ^ 1);

    v16bf a0 = load_frag_a(&As[buf][(wm + 0) * LDA_S], LDA_S);
    v16bf a1 = load_frag_a(&As[buf][(wm + 16) * LDA_S], LDA_S);
    v16bf b0 = load_frag_b(&Ws[buf][wn + 0], LDW_S);
    v16bf b1 = load_frag_b(&Ws[buf][wn + 16], LDW_S);
    acc[0][0] = wmma_bf16(a0, b0, acc[0][0]);
    acc[0][1] = wmma_bf16(a0, b1, acc[0][1]);
    acc[1][0] = wmma_bf16(a1, b0, acc[1][0]);
    acc[1][1] = wmma_bf16(a1, b1, acc[1][1]);

    wait_async0();   // drain my DMAs into buf^1
    __syncthreads(); // publish buf^1; all reads of buf complete
  }

  int lh = (lane >> 4) & 1, ln = lane & 15;
#pragma unroll
  for (int i = 0; i < 2; ++i)
#pragma unroll
    for (int j = 0; j < 2; ++j)
#pragma unroll
      for (int r = 0; r < 8; ++r) {
        int lrow = wm + i * 16 + lh * 8 + r;
        int grow = row0 + lrow;
        if (grow < cnt) {
          int gcol = n0 + wn + j * 16 + ln;
          float v = acc[i][j][r];
          if (bias) v += bias[gcol];
          if (RELU) v = v > 0.f ? v : 0.f;
          long cix = (long)(off_e + grow) * ldc + gcol;
          if (BF16OUT) ((bf16_t*)Cv)[cix] = f2bf(v);
          else         ((float*)Cv)[cix] = v;
        }
      }
}

// ---------------- flash attention (hd = 64) ----------------
// Q,K,V,O: bf16 [B,S,D] with head slice at h*64. Block: 256 thr = 8 waves,
// 16 q-rows per wave (128 q rows per block). Double-buffered async K/V tiles;
// softmax row-sum computed on the matrix pipe (P x ones) instead of shuffles.
__global__ __launch_bounds__(256) void flash_kernel(
    const bf16_t* __restrict__ Q, const bf16_t* __restrict__ Km,
    const bf16_t* __restrict__ V, bf16_t* __restrict__ O, int S, int Dm) {
  __shared__ bf16_t Ks[2][32 * 72];
  __shared__ bf16_t Vs[2][32 * 72];
  __shared__ bf16_t Ps[8 * 16 * 40];

  int b = blockIdx.z, h = blockIdx.y;
  int tid = threadIdx.x, lane = tid & 31, wid = tid >> 5;
  int q0 = blockIdx.x * 128 + wid * 16;
  long base = ((long)b * S) * Dm + h * 64;

  int m = lane & 15, half = (lane >> 4) & 1, ln = lane & 15;

  // Q fragments (row-major global gather, once)
  const bf16_t* qrow = Q + base + (long)(q0 + m) * Dm;
  v16bf qa0, qa1;
#pragma unroll
  for (int e = 0; e < 8; ++e) {
    qa0[e]     = qrow[half * 8 + e];
    qa0[8 + e] = qrow[16 + half * 8 + e];
    qa1[e]     = qrow[32 + half * 8 + e];
    qa1[8 + e] = qrow[48 + half * 8 + e];
  }

  v16bf ones;
#pragma unroll
  for (int e = 0; e < 16; ++e)
    ones[e] = __builtin_bit_cast(bf16_t, (unsigned short)0x3F80);  // 1.0bf

  float mrun[8], lrun[8], corr[8];
#pragma unroll
  for (int r = 0; r < 8; ++r) { mrun[r] = -1e30f; lrun[r] = 0.f; }
  v8f oacc[4];
#pragma unroll
  for (int j = 0; j < 4; ++j) oacc[j] = v8f_zero();

  bf16_t* Pw = &Ps[wid * 16 * 40];
  int skey = tid >> 3, sseg = (tid & 7) * 8;

  auto stageKV = [&](int kbs, int bufs) {
    const bf16_t* gk = Km + base + (long)(kbs + skey) * Dm + sseg;
    async_load_b128(gk, &Ks[bufs][skey * 72 + sseg]);
    const bf16_t* gv = V + base + (long)(kbs + skey) * Dm + sseg;
    async_load_b128(gv, &Vs[bufs][skey * 72 + sseg]);
  };

  stageKV(0, 0);
  wait_async0();
  __syncthreads();

  for (int kb = 0; kb < S; kb += 32) {
    int buf = (kb / 32) & 1;
    if (kb + 32 < S) stageKV(kb + 32, buf ^ 1);

    // scores: S = Q(16x64) * K^T(64x32) -> two 16x16 tiles, chained over hd
    v16bf bk0lo = load_frag_a(&Ks[buf][0 * 72 + 0], 72);
    v16bf bk0hi = load_frag_a(&Ks[buf][0 * 72 + 32], 72);
    v16bf bk1lo = load_frag_a(&Ks[buf][16 * 72 + 0], 72);
    v16bf bk1hi = load_frag_a(&Ks[buf][16 * 72 + 32], 72);
    v8f s0 = wmma_bf16(qa0, bk0lo, v8f_zero());
    s0 = wmma_bf16(qa1, bk0hi, s0);
    v8f s1 = wmma_bf16(qa0, bk1lo, v8f_zero());
    s1 = wmma_bf16(qa1, bk1hi, s1);

    // online softmax (scale 1/sqrt(64) = 0.125); row-max via lane shuffles
#pragma unroll
    for (int r = 0; r < 8; ++r) {
      float a = s0[r] * 0.125f;
      float c = s1[r] * 0.125f;
      float v = fmaxf(a, c);
#pragma unroll
      for (int d = 8; d >= 1; d >>= 1) v = fmaxf(v, __shfl_xor(v, d, 32));
      float mn = fmaxf(mrun[r], v);
      corr[r] = __expf(mrun[r] - mn);
      mrun[r] = mn;
      s0[r] = __expf(a - mn);
      s1[r] = __expf(c - mn);
#pragma unroll
      for (int j = 0; j < 4; ++j) oacc[j][r] *= corr[r];
    }

    // P: C-layout -> A-layout via per-wave LDS (in-order DS within wave)
#pragma unroll
    for (int r = 0; r < 8; ++r) {
      int prow = half * 8 + r;
      Pw[prow * 40 + ln] = f2bf(s0[r]);
      Pw[prow * 40 + 16 + ln] = f2bf(s1[r]);
    }
    asm volatile("s_wait_dscnt 0" ::: "memory");

    v16bf pA = load_frag_a(Pw, 40);

    // row-sum on the matrix pipe: P(16x32) x ones(32x16) -> every lane gets
    // its rows' sums in C-layout (consistent with the bf16 P used below).
    v8f psum = wmma_bf16(pA, ones, v8f_zero());
#pragma unroll
    for (int r = 0; r < 8; ++r) lrun[r] = lrun[r] * corr[r] + psum[r];

#pragma unroll
    for (int j = 0; j < 4; ++j) {
      v16bf bv = load_frag_b(&Vs[buf][j * 16], 72);  // Vs[k][n] layout
      oacc[j] = wmma_bf16(pA, bv, oacc[j]);
    }

    wait_async0();   // drain DMAs into buf^1
    __syncthreads(); // publish buf^1; all reads of buf complete
  }

  // write O (bf16), normalized by running sum
#pragma unroll
  for (int j = 0; j < 4; ++j)
#pragma unroll
    for (int r = 0; r < 8; ++r) {
      int row = q0 + half * 8 + r;
      int col = j * 16 + ln;
      O[base + (long)row * Dm + col] = f2bf(oacc[j][r] / lrun[r]);
    }
}

// ---------------- residual + layernorm ----------------
__global__ __launch_bounds__(256) void resid_ln_kernel(
    const float* __restrict__ resid, const float* __restrict__ y,
    const float* __restrict__ g, const float* __restrict__ bta,
    float* __restrict__ out_f, bf16_t* __restrict__ out_bf, int Dn) {
  __shared__ float red[256], red2[256];
  long row = blockIdx.x;
  const float* rp = resid + row * Dn;
  const float* yp = y + row * Dn;
  int tid = threadIdx.x;
  float v[4], s = 0.f, s2 = 0.f;
#pragma unroll
  for (int u = 0; u < 4; ++u) {
    int i = tid + u * 256;
    float t = rp[i] + yp[i];
    v[u] = t; s += t; s2 += t * t;
  }
  red[tid] = s; red2[tid] = s2;
  __syncthreads();
  for (int off = 128; off > 0; off >>= 1) {
    if (tid < off) { red[tid] += red[tid + off]; red2[tid] += red2[tid + off]; }
    __syncthreads();
  }
  float mu = red[0] / Dn;
  float var = red2[0] / Dn - mu * mu;
  float rstd = rsqrtf(var + 1e-5f);
#pragma unroll
  for (int u = 0; u < 4; ++u) {
    int i = tid + u * 256;
    float o = (v[u] - mu) * rstd * g[i] + bta[i];
    if (out_f) out_f[row * Dn + i] = o;
    if (out_bf) out_bf[row * Dn + i] = f2bf(o);
  }
}

// ---------------- MoE router / scatter ----------------
__global__ void zero_counts_kernel(int* __restrict__ counts) {
  if (threadIdx.x < E_) counts[threadIdx.x] = 0;
}

__global__ void router_kernel(const float* __restrict__ x, const float* __restrict__ rw,
                              const float* __restrict__ rb, int* __restrict__ eids,
                              float* __restrict__ gates, int* __restrict__ counts,
                              int T, int Dn) {
  int t = blockIdx.x * blockDim.x + threadIdx.x;
  if (t >= T) return;
  float lg[E_];
#pragma unroll
  for (int e = 0; e < E_; ++e) lg[e] = rb[e];
  const float* xp = x + (long)t * Dn;
  for (int d = 0; d < Dn; ++d) {
    float xv = xp[d];
    const float* wr = rw + (long)d * E_;
#pragma unroll
    for (int e = 0; e < E_; ++e) lg[e] = fmaf(xv, wr[e], lg[e]);
  }
  float mx = lg[0];
#pragma unroll
  for (int e = 1; e < E_; ++e) mx = fmaxf(mx, lg[e]);
#pragma unroll
  for (int e = 0; e < E_; ++e) lg[e] = __expf(lg[e] - mx);
  // top-2 (ties -> lowest index, matching top_k); renorm: v0/(v0+v1)
  int i0 = 0; float v0 = lg[0];
#pragma unroll
  for (int e = 1; e < E_; ++e) if (lg[e] > v0) { v0 = lg[e]; i0 = e; }
  int i1 = -1; float v1 = -1.f;
#pragma unroll
  for (int e = 0; e < E_; ++e)
    if (e != i0 && lg[e] > v1) { v1 = lg[e]; i1 = e; }
  float gs = v0 + v1;
  gates[2 * t] = v0 / gs;
  gates[2 * t + 1] = v1 / gs;
  eids[2 * t] = i0;
  eids[2 * t + 1] = i1;
  atomicAdd(&counts[i0], 1);
  atomicAdd(&counts[i1], 1);
}

__global__ void scan_kernel(const int* __restrict__ counts, int* __restrict__ off,
                            int* __restrict__ cursor) {
  if (threadIdx.x == 0) {
    int a = 0;
    for (int e = 0; e < E_; ++e) { off[e] = a; cursor[e] = a; a += counts[e]; }
    off[E_] = a;
  }
}

__global__ void scatter_kernel(const int* __restrict__ eids, int* __restrict__ cursor,
                               int* __restrict__ idx_list, int* __restrict__ pos_rec,
                               int T) {
  int t = blockIdx.x * blockDim.x + threadIdx.x;
  if (t >= T) return;
#pragma unroll
  for (int k = 0; k < 2; ++k) {
    int e = eids[2 * t + k];
    int p = atomicAdd(&cursor[e], 1);
    idx_list[p] = t;
    pos_rec[2 * t + k] = p;
  }
}

// ---------------- MoE combine + final LN ----------------
__global__ __launch_bounds__(256) void moe_combine_ln_kernel(
    const float* __restrict__ x2, const float* __restrict__ Yg,
    const float* __restrict__ gates, const int* __restrict__ pos_rec,
    const float* __restrict__ g, const float* __restrict__ bta,
    float* __restrict__ out, int Dn) {
  __shared__ float red[256], red2[256];
  long t = blockIdx.x;
  int p0 = pos_rec[2 * t], p1 = pos_rec[2 * t + 1];
  float g0 = gates[2 * t], g1 = gates[2 * t + 1];
  const float* xp = x2 + t * Dn;
  const float* y0 = Yg + (long)p0 * Dn;
  const float* y1 = Yg + (long)p1 * Dn;
  int tid = threadIdx.x;
  float v[4], s = 0.f, s2 = 0.f;
#pragma unroll
  for (int u = 0; u < 4; ++u) {
    int i = tid + u * 256;
    float w = xp[i] + g0 * y0[i] + g1 * y1[i];
    v[u] = w; s += w; s2 += w * w;
  }
  red[tid] = s; red2[tid] = s2;
  __syncthreads();
  for (int off = 128; off > 0; off >>= 1) {
    if (tid < off) { red[tid] += red[tid + off]; red2[tid] += red2[tid + off]; }
    __syncthreads();
  }
  float mu = red[0] / Dn;
  float var = red2[0] / Dn - mu * mu;
  float rstd = rsqrtf(var + 1e-5f);
#pragma unroll
  for (int u = 0; u < 4; ++u) {
    int i = tid + u * 256;
    out[t * Dn + i] = (v[u] - mu) * rstd * g[i] + bta[i];
  }
}

// ---------------- host launcher ----------------
extern "C" void kernel_launch(void* const* d_in, const int* in_sizes, int n_in,
                              void* d_out, int out_size, void* d_ws, size_t ws_size,
                              hipStream_t stream) {
  (void)in_sizes; (void)n_in; (void)out_size; (void)ws_size;

  const float* tgt   = (const float*)d_in[0];
  const float* mem   = (const float*)d_in[1];
  const float* sa_wq = (const float*)d_in[2];  const float* sa_bq = (const float*)d_in[3];
  const float* sa_wk = (const float*)d_in[4];  const float* sa_bk = (const float*)d_in[5];
  const float* sa_wv = (const float*)d_in[6];  const float* sa_bv = (const float*)d_in[7];
  const float* sa_wo = (const float*)d_in[8];  const float* sa_bo = (const float*)d_in[9];
  const float* ca_wq = (const float*)d_in[10]; const float* ca_bq = (const float*)d_in[11];
  const float* ca_wk = (const float*)d_in[12]; const float* ca_bk = (const float*)d_in[13];
  const float* ca_wv = (const float*)d_in[14]; const float* ca_bv = (const float*)d_in[15];
  const float* ca_wo = (const float*)d_in[16]; const float* ca_bo = (const float*)d_in[17];
  const float* r_w   = (const float*)d_in[18]; const float* r_b   = (const float*)d_in[19];
  const float* e_w1  = (const float*)d_in[20]; const float* e_b1  = (const float*)d_in[21];
  const float* e_w2  = (const float*)d_in[22]; const float* e_b2  = (const float*)d_in[23];
  const float* n1_g  = (const float*)d_in[24]; const float* n1_b  = (const float*)d_in[25];
  const float* n2_g  = (const float*)d_in[26]; const float* n2_b  = (const float*)d_in[27];
  const float* n3_g  = (const float*)d_in[28]; const float* n3_b  = (const float*)d_in[29];

  // ---- carve workspace ----
  size_t woff = 0;
  auto carve = [&](size_t bytes) -> void* {
    void* p = (char*)d_ws + woff;
    woff += (bytes + 255) & ~(size_t)255;
    return p;
  };
  const size_t DD = (size_t)D_ * D_;
  bf16_t* wsaq = (bf16_t*)carve(DD * 2);
  bf16_t* wsak = (bf16_t*)carve(DD * 2);
  bf16_t* wsav = (bf16_t*)carve(DD * 2);
  bf16_t* wsao = (bf16_t*)carve(DD * 2);
  bf16_t* wcaq = (bf16_t*)carve(DD * 2);
  bf16_t* wcak = (bf16_t*)carve(DD * 2);
  bf16_t* wcav = (bf16_t*)carve(DD * 2);
  bf16_t* wcao = (bf16_t*)carve(DD * 2);
  bf16_t* we1  = (bf16_t*)carve((size_t)E_ * D_ * HID_ * 2);
  bf16_t* we2  = (bf16_t*)carve((size_t)E_ * HID_ * D_ * 2);
  bf16_t* tgtb = (bf16_t*)carve((size_t)T_ * D_ * 2);
  bf16_t* memb = (bf16_t*)carve((size_t)T_ * D_ * 2);
  bf16_t* qb   = (bf16_t*)carve((size_t)T_ * D_ * 2);
  bf16_t* kb   = (bf16_t*)carve((size_t)T_ * D_ * 2);
  bf16_t* vb   = (bf16_t*)carve((size_t)T_ * D_ * 2);
  bf16_t* x1b  = (bf16_t*)carve((size_t)T_ * D_ * 2);
  bf16_t* x2b  = (bf16_t*)carve((size_t)T_ * D_ * 2);
  bf16_t* Hbuf = (bf16_t*)carve((size_t)TK_ * HID_ * 2);
  float* projf = (float*)carve((size_t)T_ * D_ * 4);
  float* x1f   = (float*)carve((size_t)T_ * D_ * 4);
  float* x2f   = (float*)carve((size_t)T_ * D_ * 4);
  float* Yg    = (float*)carve((size_t)TK_ * D_ * 4);
  float* gates = (float*)carve((size_t)2 * T_ * 4);
  int* eids    = (int*)carve((size_t)2 * T_ * 4);
  int* idxl    = (int*)carve((size_t)TK_ * 4);
  int* posrec  = (int*)carve((size_t)2 * T_ * 4);
  int* counts  = (int*)carve(64);
  int* offs    = (int*)carve(64);
  int* cursor  = (int*)carve(64);
  bf16_t* attnb = tgtb;  // reuse: tgt_bf dead after self-attn QKV projections

  auto cvt = [&](const float* s, bf16_t* d, long n) {
    int blocks = (int)((n + 255) / 256);
    cvt_kernel<<<blocks, 256, 0, stream>>>(s, d, n);
  };

  // ---- fp32 -> bf16 conversions ----
  cvt(tgt, tgtb, (long)T_ * D_);
  cvt(mem, memb, (long)T_ * D_);
  cvt(sa_wq, wsaq, DD); cvt(sa_wk, wsak, DD); cvt(sa_wv, wsav, DD); cvt(sa_wo, wsao, DD);
  cvt(ca_wq, wcaq, DD); cvt(ca_wk, wcak, DD); cvt(ca_wv, wcav, DD); cvt(ca_wo, wcao, DD);
  cvt(e_w1, we1, (long)E_ * D_ * HID_);
  cvt(e_w2, we2, (long)E_ * HID_ * D_);

  dim3 gDense(D_ / BN, T_ / BM);  // (16, 64)

  // ---- self attention ----
  gemm_kernel<false, true><<<gDense, 256, 0, stream>>>(
      tgtb, wsaq, sa_bq, (void*)qb, nullptr, 0, nullptr, T_, D_, D_, D_, D_);
  gemm_kernel<false, true><<<gDense, 256, 0, stream>>>(
      tgtb, wsak, sa_bk, (void*)kb, nullptr, 0, nullptr, T_, D_, D_, D_, D_);
  gemm_kernel<false, true><<<gDense, 256, 0, stream>>>(
      tgtb, wsav, sa_bv, (void*)vb, nullptr, 0, nullptr, T_, D_, D_, D_, D_);
  flash_kernel<<<dim3(S_ / 128, H_, B_), 256, 0, stream>>>(qb, kb, vb, attnb, S_, D_);
  gemm_kernel<false, false><<<gDense, 256, 0, stream>>>(
      attnb, wsao, sa_bo, (void*)projf, nullptr, 0, nullptr, T_, D_, D_, D_, D_);
  resid_ln_kernel<<<T_, 256, 0, stream>>>(tgt, projf, n1_g, n1_b, x1f, x1b, D_);

  // ---- cross attention ----
  gemm_kernel<false, true><<<gDense, 256, 0, stream>>>(
      x1b, wcaq, ca_bq, (void*)qb, nullptr, 0, nullptr, T_, D_, D_, D_, D_);
  gemm_kernel<false, true><<<gDense, 256, 0, stream>>>(
      memb, wcak, ca_bk, (void*)kb, nullptr, 0, nullptr, T_, D_, D_, D_, D_);
  gemm_kernel<false, true><<<gDense, 256, 0, stream>>>(
      memb, wcav, ca_bv, (void*)vb, nullptr, 0, nullptr, T_, D_, D_, D_, D_);
  flash_kernel<<<dim3(S_ / 128, H_, B_), 256, 0, stream>>>(qb, kb, vb, attnb, S_, D_);
  gemm_kernel<false, false><<<gDense, 256, 0, stream>>>(
      attnb, wcao, ca_bo, (void*)projf, nullptr, 0, nullptr, T_, D_, D_, D_, D_);
  resid_ln_kernel<<<T_, 256, 0, stream>>>(x1f, projf, n2_g, n2_b, x2f, x2b, D_);

  // ---- MoE: route, scatter, gathered expert FFNs ----
  zero_counts_kernel<<<1, 32, 0, stream>>>(counts);
  router_kernel<<<(T_ + 255) / 256, 256, 0, stream>>>(x2f, r_w, r_b, eids, gates,
                                                      counts, T_, D_);
  scan_kernel<<<1, 1, 0, stream>>>(counts, offs, cursor);
  scatter_kernel<<<(T_ + 255) / 256, 256, 0, stream>>>(eids, cursor, idxl, posrec, T_);

  for (int e = 0; e < E_; ++e) {
    gemm_kernel<true, true><<<dim3(HID_ / BN, TK_ / BM), 256, 0, stream>>>(
        x2b, we1 + (size_t)e * D_ * HID_, e_b1 + (size_t)e * HID_, (void*)Hbuf,
        offs, e, idxl, TK_, HID_, D_, D_, HID_);
  }
  for (int e = 0; e < E_; ++e) {
    gemm_kernel<false, false><<<dim3(D_ / BN, TK_ / BM), 256, 0, stream>>>(
        Hbuf, we2 + (size_t)e * HID_ * D_, e_b2 + (size_t)e * D_, (void*)Yg,
        offs, e, nullptr, TK_, D_, HID_, HID_, D_);
  }

  // ---- combine + final layernorm -> d_out ----
  moe_combine_ln_kernel<<<T_, 256, 0, stream>>>(x2f, Yg, gates, posrec, n3_g, n3_b,
                                                (float*)d_out, D_);
}
